// GRU_update_2345052144197
// MI455X (gfx1250) — matrix operand order repaired and checked by reading, not compiled, not run on hardware
//
#include <hip/hip_runtime.h>

// ---------------- problem sizes ----------------
#define BATCH   16384
#define HID     512
#define OUTD    4
#define FFD     2048
#define HOR     5

// ---------------- GEMM tiling ----------------
#define BM 128           // workgroup rows
#define BN 64            // workgroup cols
#define BK 64            // K per LDS stage (2 WMMA k-chunks)
#define LDK (BK + 8)     // padded LDS row stride in halves: 144B = 9*16B, 36-bank stride

typedef __attribute__((ext_vector_type(16))) _Float16 v16h;
typedef __attribute__((ext_vector_type(8)))  float    v8f;

// ---- fragment loaders per CDNA5 WMMA VGPR layouts (CDNA5 ISA) ----
__device__ __forceinline__ v16h load_frag_a(const _Float16* __restrict__ base, int ldr) {
  const int lane  = threadIdx.x & 31;
  const int m     = lane & 15;
  const int kbase = (lane >> 4) ? 8 : 0;
  v16h a;
#pragma unroll
  for (int v = 0; v < 8; ++v) {
    const int k0 = kbase + ((v < 4) ? (2 * v) : (16 + 2 * (v - 4)));
    a[2 * v]     = base[m * ldr + k0];
    a[2 * v + 1] = base[m * ldr + k0 + 1];
  }
  return a;
}

__device__ __forceinline__ v16h load_frag_b(const _Float16* __restrict__ base, int ldr) {
  const int lane  = threadIdx.x & 31;
  const int n     = lane & 15;
  const int kbase = (lane >> 4) ? 8 : 0;
  v16h b;
#pragma unroll
  for (int v = 0; v < 8; ++v) {
    const int k0 = kbase + ((v < 4) ? (2 * v) : (16 + 2 * (v - 4)));
    b[2 * v]     = base[n * ldr + k0];
    b[2 * v + 1] = base[n * ldr + k0 + 1];
  }
  return b;
}

// CDNA5 async global->LDS 16B/lane copy (GLOBAL_LOAD_ASYNC_TO_LDS_B128, ASYNCcnt)
__device__ __forceinline__ void async_copy_b128(unsigned lds_addr, unsigned long long gaddr) {
  asm volatile("global_load_async_to_lds_b128 %0, %1, off"
               :: "v"(lds_addr), "v"(gaddr) : "memory");
}
// in-order completion => waiting ASYNCcnt<=N retires the oldest ops first
template <int N>
__device__ __forceinline__ void wait_async_le() {
  asm volatile("s_wait_asynccnt %0" :: "i"(N) : "memory");
}

// low 32 bits of a generic pointer to __shared__ == LDS byte address (aperture truncation)
__device__ __forceinline__ unsigned lds_addr_of(const void* p) {
  return (unsigned)(size_t)p;
}

// ---------------- generic LDS-staged, double-buffered WMMA GEMM core ----------------
__device__ __forceinline__ void gemm_tile_async(const _Float16* __restrict__ A, int lda,
                                                const _Float16* __restrict__ W, int ldw,
                                                int K, int row0,
                                                _Float16 (*sA)[BM * LDK],
                                                _Float16 (*sB)[BN * LDK],
                                                v8f acc[2][2]) {
  const int tid  = threadIdx.x;
  const int wave = tid >> 5;
  const int wm   = wave & 3;
  const int wn   = wave >> 2;
  const v8f vz = {0.f, 0.f, 0.f, 0.f, 0.f, 0.f, 0.f, 0.f};
#pragma unroll
  for (int i = 0; i < 2; ++i)
#pragma unroll
    for (int j = 0; j < 2; ++j) acc[i][j] = vz;

  // per-thread staging offsets: 16B chunks, 8 chunks per BK=64 row
  unsigned relA[4]; unsigned long long gA[4];
#pragma unroll
  for (int c = 0; c < 4; ++c) {
    const int e = tid + c * 256, r = e >> 3, kc = e & 7;
    relA[c] = r * (LDK * 2) + kc * 16;
    gA[c]   = (unsigned long long)(A + (size_t)(row0 + r) * lda + kc * 8);
  }
  unsigned relW[2]; unsigned long long gW[2];
#pragma unroll
  for (int c = 0; c < 2; ++c) {
    const int e = tid + c * 256, r = e >> 3, kc = e & 7;
    relW[c] = r * (LDK * 2) + kc * 16;
    gW[c]   = (unsigned long long)(W + (size_t)r * ldw + kc * 8);
  }

  auto issue = [&](int k0, int buf) {
    const unsigned long long koff = (unsigned long long)k0 * 2;
    const unsigned baseA = lds_addr_of(&sA[buf][0]);
    const unsigned baseB = lds_addr_of(&sB[buf][0]);
#pragma unroll
    for (int c = 0; c < 4; ++c) async_copy_b128(baseA + relA[c], gA[c] + koff);
#pragma unroll
    for (int c = 0; c < 2; ++c) async_copy_b128(baseB + relW[c], gW[c] + koff);
  };

  issue(0, 0);
  int buf = 0;
  for (int k0 = 0; k0 < K; k0 += BK, buf ^= 1) {
    __syncthreads();  // all waves done reading buf^1
    if (k0 + BK < K) { issue(k0 + BK, buf ^ 1); wait_async_le<6>(); }
    else             { wait_async_le<0>(); }
    __syncthreads();  // stage data visible to all waves
#pragma unroll
    for (int kk = 0; kk < BK; kk += 32) {
      const v16h a0 = load_frag_a(&sA[buf][0] + (wm * 32 + 0)  * LDK + kk, LDK);
      const v16h a1 = load_frag_a(&sA[buf][0] + (wm * 32 + 16) * LDK + kk, LDK);
      const v16h b0 = load_frag_b(&sB[buf][0] + (wn * 32 + 0)  * LDK + kk, LDK);
      const v16h b1 = load_frag_b(&sB[buf][0] + (wn * 32 + 16) * LDK + kk, LDK);
      acc[0][0] = __builtin_amdgcn_wmma_f32_16x16x32_f16(false, a0, false, b0, (short)0, acc[0][0], false, false);
      acc[0][1] = __builtin_amdgcn_wmma_f32_16x16x32_f16(false, a0, false, b1, (short)0, acc[0][1], false, false);
      acc[1][0] = __builtin_amdgcn_wmma_f32_16x16x32_f16(false, a1, false, b0, (short)0, acc[1][0], false, false);
      acc[1][1] = __builtin_amdgcn_wmma_f32_16x16x32_f16(false, a1, false, b1, (short)0, acc[1][1], false, false);
    }
  }
}

// ---------------- fused GRU gate kernel (double-buffered, 3 gates/K-loop) ----------------
__global__ __launch_bounds__(256) void gru_gates_kernel(
    const _Float16* __restrict__ A, int lda,
    const _Float16* __restrict__ h_in, int ldh,
    const float* __restrict__ xy_src, int xy_rs,
    const _Float16* __restrict__ w_hh16,            // [3*HID, HID] f16
    const float* __restrict__ w_ih,                 // [3*HID, OUTD] f32
    const float* __restrict__ b_ih, const float* __restrict__ b_hh,
    _Float16* __restrict__ out_h, int ldo) {
  __shared__ _Float16 sA[2][BM * LDK] __attribute__((aligned(16)));
  __shared__ _Float16 sB[2][3 * BN * LDK] __attribute__((aligned(16)));
  const int row0 = blockIdx.x * BM;
  const int col0 = blockIdx.y * BN;
  const int tid  = threadIdx.x;
  const int lane = tid & 31;
  const int wave = tid >> 5;
  const int wm = wave & 3, wn = wave >> 2;

  v8f acc[3][2][2];
  const v8f vz = {0.f, 0.f, 0.f, 0.f, 0.f, 0.f, 0.f, 0.f};
#pragma unroll
  for (int g = 0; g < 3; ++g)
#pragma unroll
    for (int i = 0; i < 2; ++i)
#pragma unroll
      for (int j = 0; j < 2; ++j) acc[g][i][j] = vz;

  unsigned relA[4]; unsigned long long gA[4];
#pragma unroll
  for (int c = 0; c < 4; ++c) {
    const int e = tid + c * 256, r = e >> 3, kc = e & 7;
    relA[c] = r * (LDK * 2) + kc * 16;
    gA[c]   = (unsigned long long)(A + (size_t)(row0 + r) * lda + kc * 8);
  }
  unsigned relW[3][2]; unsigned long long gW[3][2];
#pragma unroll
  for (int g = 0; g < 3; ++g)
#pragma unroll
    for (int c = 0; c < 2; ++c) {
      const int e = tid + c * 256, r = e >> 3, kc = e & 7;
      relW[g][c] = (g * BN + r) * (LDK * 2) + kc * 16;
      gW[g][c]   = (unsigned long long)(w_hh16 + (size_t)(g * HID + col0 + r) * HID + kc * 8);
    }

  auto issue = [&](int k0, int buf) {
    const unsigned long long koff = (unsigned long long)k0 * 2;
    const unsigned baseA = lds_addr_of(&sA[buf][0]);
    const unsigned baseB = lds_addr_of(&sB[buf][0]);
#pragma unroll
    for (int c = 0; c < 4; ++c) async_copy_b128(baseA + relA[c], gA[c] + koff);
#pragma unroll
    for (int g = 0; g < 3; ++g)
#pragma unroll
      for (int c = 0; c < 2; ++c) async_copy_b128(baseB + relW[g][c], gW[g][c] + koff);
  };

  issue(0, 0);
  int buf = 0;
  for (int k0 = 0; k0 < HID; k0 += BK, buf ^= 1) {
    __syncthreads();
    if (k0 + BK < HID) { issue(k0 + BK, buf ^ 1); wait_async_le<10>(); }
    else               { wait_async_le<0>(); }
    __syncthreads();
#pragma unroll
    for (int kk = 0; kk < BK; kk += 32) {
      const v16h a0 = load_frag_a(&sA[buf][0] + (wm * 32 + 0)  * LDK + kk, LDK);
      const v16h a1 = load_frag_a(&sA[buf][0] + (wm * 32 + 16) * LDK + kk, LDK);
#pragma unroll
      for (int g = 0; g < 3; ++g) {
        const v16h b0 = load_frag_b(&sB[buf][0] + (g * BN + wn * 32 + 0)  * LDK + kk, LDK);
        const v16h b1 = load_frag_b(&sB[buf][0] + (g * BN + wn * 32 + 16) * LDK + kk, LDK);
        acc[g][0][0] = __builtin_amdgcn_wmma_f32_16x16x32_f16(false, a0, false, b0, (short)0, acc[g][0][0], false, false);
        acc[g][0][1] = __builtin_amdgcn_wmma_f32_16x16x32_f16(false, a0, false, b1, (short)0, acc[g][0][1], false, false);
        acc[g][1][0] = __builtin_amdgcn_wmma_f32_16x16x32_f16(false, a1, false, b0, (short)0, acc[g][1][0], false, false);
        acc[g][1][1] = __builtin_amdgcn_wmma_f32_16x16x32_f16(false, a1, false, b1, (short)0, acc[g][1][1], false, false);
      }
    }
  }

#pragma unroll
  for (int j = 0; j < 2; ++j) {
    const int n = col0 + wn * 32 + j * 16 + (lane & 15);
    const float bir = b_ih[n],            bhr = b_hh[n];
    const float biz = b_ih[HID + n],      bhz = b_hh[HID + n];
    const float bin_ = b_ih[2 * HID + n], bhn = b_hh[2 * HID + n];
    const float* wr = w_ih + (size_t)n * OUTD;
    const float* wz = w_ih + (size_t)(HID + n) * OUTD;
    const float* wv = w_ih + (size_t)(2 * HID + n) * OUTD;
#pragma unroll
    for (int i = 0; i < 2; ++i) {
#pragma unroll
      for (int g = 0; g < 8; ++g) {
        const int m = row0 + wm * 32 + i * 16 + ((lane < 16) ? g : 8 + g);
        const float* xyp = xy_src + (size_t)m * xy_rs;
        const float x0 = xyp[0], x1 = xyp[1], x2 = xyp[2], x3 = xyp[3];
        const float gr = acc[0][i][j][g] + bhr + bir + x0 * wr[0] + x1 * wr[1] + x2 * wr[2] + x3 * wr[3];
        const float gz = acc[1][i][j][g] + bhz + biz + x0 * wz[0] + x1 * wz[1] + x2 * wz[2] + x3 * wz[3];
        const float hn = acc[2][i][j][g] + bhn;
        const float gn = bin_ + x0 * wv[0] + x1 * wv[1] + x2 * wv[2] + x3 * wv[3];
        const float r  = 1.f / (1.f + __expf(-gr));
        const float z  = 1.f / (1.f + __expf(-gz));
        const float nn = tanhf(gn + r * hn);
        const float hp = (float)h_in[(size_t)m * ldh + n];
        out_h[(size_t)m * ldo + n] = (_Float16)((1.f - z) * nn + z * hp);
      }
    }
  }
}

// ---------------- generic linear (+bias, optional relu / relu'd copy) ----------------
__global__ __launch_bounds__(256) void linear_kernel(
    const _Float16* __restrict__ A, int lda,
    const _Float16* __restrict__ W, int ldw, int K,
    const float* __restrict__ bias,
    _Float16* __restrict__ out, int ldo, int relu_out,
    _Float16* __restrict__ out2, int ldo2) {     // out2: relu(v) copy (may be null)
  __shared__ _Float16 sA[2][BM * LDK] __attribute__((aligned(16)));
  __shared__ _Float16 sB[2][BN * LDK] __attribute__((aligned(16)));
  const int row0 = blockIdx.x * BM;
  const int col0 = blockIdx.y * BN;
  v8f acc[2][2];
  gemm_tile_async(A, lda, W + (size_t)col0 * ldw, ldw, K, row0, sA, sB, acc);
  const int lane = threadIdx.x & 31;
  const int wave = threadIdx.x >> 5;
  const int wm = wave & 3, wn = wave >> 2;
#pragma unroll
  for (int j = 0; j < 2; ++j) {
    const int n = col0 + wn * 32 + j * 16 + (lane & 15);
    const float bn = bias[n];
#pragma unroll
    for (int i = 0; i < 2; ++i) {
#pragma unroll
      for (int g = 0; g < 8; ++g) {
        const int m = row0 + wm * 32 + i * 16 + ((lane < 16) ? g : 8 + g);
        float v = acc[i][j][g] + bn;
        const float vr = fmaxf(v, 0.f);
        if (relu_out) v = vr;
        out[(size_t)m * ldo + n] = (_Float16)v;
        if (out2) out2[(size_t)m * ldo2 + n] = (_Float16)vr;
      }
    }
  }
}

// ---------------- mlp tail: xy += t @ w2^T + b2 ; out[:,step,:] = xy ----------------
__global__ __launch_bounds__(256) void mlp2_kernel(
    const _Float16* __restrict__ t, const float* __restrict__ w2,
    const float* __restrict__ b2, float* __restrict__ xy,
    float* __restrict__ out_step) {
  const int b    = (blockIdx.x * blockDim.x + threadIdx.x) >> 5;
  const int lane = threadIdx.x & 31;
  const _Float16* tr = t + (size_t)b * FFD;
  float a0 = 0.f, a1 = 0.f, a2 = 0.f, a3 = 0.f;
  for (int k2 = lane; k2 < FFD / 2; k2 += 32) {
    union { unsigned u; _Float16 h[2]; } p;
    p.u = *(const unsigned*)(tr + 2 * k2);
    const float t0 = (float)p.h[0], t1 = (float)p.h[1];
    const int k = 2 * k2;
    a0 += t0 * w2[0 * FFD + k] + t1 * w2[0 * FFD + k + 1];
    a1 += t0 * w2[1 * FFD + k] + t1 * w2[1 * FFD + k + 1];
    a2 += t0 * w2[2 * FFD + k] + t1 * w2[2 * FFD + k + 1];
    a3 += t0 * w2[3 * FFD + k] + t1 * w2[3 * FFD + k + 1];
  }
#pragma unroll
  for (int off = 16; off > 0; off >>= 1) {
    a0 += __shfl_xor(a0, off);
    a1 += __shfl_xor(a1, off);
    a2 += __shfl_xor(a2, off);
    a3 += __shfl_xor(a3, off);
  }
  if (lane == 0) {
    float* xyp = xy + (size_t)b * OUTD;
    float* op  = out_step + (size_t)b * (HOR * OUTD);
    const float v0 = xyp[0] + a0 + b2[0];
    const float v1 = xyp[1] + a1 + b2[1];
    const float v2 = xyp[2] + a2 + b2[2];
    const float v3 = xyp[3] + a3 + b2[3];
    xyp[0] = v0; xyp[1] = v1; xyp[2] = v2; xyp[3] = v3;
    op[0] = v0; op[1] = v1; op[2] = v2; op[3] = v3;
  }
}

// ---------------- helpers ----------------
__global__ void cvt_f32_f16_kernel(const float* __restrict__ src,
                                   _Float16* __restrict__ dst, int n) {
  const int i = blockIdx.x * blockDim.x + threadIdx.x;
  if (i < n) dst[i] = (_Float16)src[i];
}

__global__ void zero_f32_kernel(float* __restrict__ p, int n) {
  const int i = blockIdx.x * blockDim.x + threadIdx.x;
  if (i < n) p[i] = 0.f;
}

// ---------------- host launcher ----------------
extern "C" void kernel_launch(void* const* d_in, const int* in_sizes, int n_in,
                              void* d_out, int out_size, void* d_ws, size_t ws_size,
                              hipStream_t stream) {
  const float* pv   = (const float*)d_in[0];   // [B, HOR, OUTD]
  const float* ptf  = (const float*)d_in[1];   // [B, HID]
  const float* w_ih = (const float*)d_in[2];   // [3H, OUTD]
  const float* w_hh = (const float*)d_in[3];   // [3H, H]
  const float* b_ih = (const float*)d_in[4];
  const float* b_hh = (const float*)d_in[5];
  const float* w_fc = (const float*)d_in[6];   // [H, 2H]
  const float* b_fc = (const float*)d_in[7];
  const float* w1   = (const float*)d_in[8];   // [FF, H]
  const float* b1   = (const float*)d_in[9];
  const float* w2   = (const float*)d_in[10];  // [OUTD, FF]
  const float* b2   = (const float*)d_in[11];
  float* out = (float*)d_out;

  char* ws = (char*)d_ws;
  size_t off = 0;
  _Float16* whh16 = (_Float16*)(ws + off); off += (size_t)3 * HID * HID * 2;
  _Float16* wfc16 = (_Float16*)(ws + off); off += (size_t)HID * 2 * HID * 2;
  _Float16* w116  = (_Float16*)(ws + off); off += (size_t)FFD * HID * 2;
  _Float16* hx16  = (_Float16*)(ws + off); off += (size_t)BATCH * HID * 2;
  _Float16* hxr16 = (_Float16*)(ws + off); off += (size_t)BATCH * HID * 2;
  _Float16* hcat  = (_Float16*)(ws + off); off += (size_t)BATCH * 2 * HID * 2;
  _Float16* t16   = (_Float16*)(ws + off); off += (size_t)BATCH * FFD * 2;
  float*    xy    = (float*)(ws + off);    off += (size_t)BATCH * OUTD * 4;

  const int T = 256;
  cvt_f32_f16_kernel<<<(3 * HID * HID + T - 1) / T, T, 0, stream>>>(w_hh, whh16, 3 * HID * HID);
  cvt_f32_f16_kernel<<<(HID * 2 * HID + T - 1) / T, T, 0, stream>>>(w_fc, wfc16, HID * 2 * HID);
  cvt_f32_f16_kernel<<<(FFD * HID + T - 1) / T, T, 0, stream>>>(w1, w116, FFD * HID);
  cvt_f32_f16_kernel<<<(BATCH * HID + T - 1) / T, T, 0, stream>>>(ptf, hx16, BATCH * HID);
  zero_f32_kernel<<<(BATCH * OUTD + T - 1) / T, T, 0, stream>>>(xy, BATCH * OUTD);

  const dim3 gGates(BATCH / BM, HID / BN);   // 128 x 8
  const dim3 gFc(BATCH / BM, HID / BN);      // 128 x 8
  const dim3 gM1(BATCH / BM, FFD / BN);      // 128 x 32

  for (int s = 0; s < HOR; ++s) {
    // h1 = GRUCell(xy, hx)      -> hcat[:, 0:H]
    gru_gates_kernel<<<gGates, T, 0, stream>>>(hx16, HID, hx16, HID, xy, OUTD,
                                               whh16, w_ih, b_ih, b_hh, hcat, 2 * HID);
    // h2 = GRUCell(pv[:,s], h1) -> hcat[:, H:2H]
    gru_gates_kernel<<<gGates, T, 0, stream>>>(hcat, 2 * HID, hcat, 2 * HID,
                                               pv + s * OUTD, HOR * OUTD,
                                               whh16, w_ih, b_ih, b_hh, hcat + HID, 2 * HID);
    // hx = hcat @ w_fc^T + b_fc ; hxr = relu(hx)
    linear_kernel<<<gFc, T, 0, stream>>>(hcat, 2 * HID, wfc16, 2 * HID, 2 * HID,
                                         b_fc, hx16, HID, 0, hxr16, HID);
    // t = relu(hxr @ w1^T + b1)
    linear_kernel<<<gM1, T, 0, stream>>>(hxr16, HID, w116, HID, HID,
                                         b1, t16, FFD, 1, (_Float16*)nullptr, 0);
    // xy += t @ w2^T + b2 ; out[:, s, :] = xy
    mlp2_kernel<<<BATCH / 8, T, 0, stream>>>(t16, w2, b2, xy, out + s * OUTD);
  }
}
